// TimesBlockTN_66520453480528
// MI455X (gfx1250) — matrix operand order repaired and checked by reading, not compiled, not run on hardware
//
#include <hip/hip_runtime.h>
#include <cstdint>
#include <cstddef>

// ---------------------------------------------------------------------------
// TimesNet block for MI455X (gfx1250).
// Gate-combined single 11x11 conv per inception block, implicit GEMM on
// v_wmma_f32_16x16x32_f16, per-tap weight slabs staged in LDS (64KB) and
// shared by 16 waves, software-pipelined against the WMMA stream.
// Working set (~100MB) fits in the 192MB L2.
// ---------------------------------------------------------------------------

typedef __attribute__((ext_vector_type(16))) _Float16 v16h;
typedef __attribute__((ext_vector_type(8)))  float    v8f;

#define T_LEN   1024
#define C_MODEL 128
#define C_FF    256
#define NBATCH  64
#define NJOBS   128      // NBATCH * K_TOP
#define RTAPS   121      // 11x11 combined kernel
#define HROWS   1056     // >= max TT (1035), hidden row stride
#define SLAB_U4 4096     // one tap slab = 128*256 halves = 64KB = 4096 uint4

// workspace layout (bytes)
#define OFF_META 0                                               // 128 * 16B {p,P,TT,w}
#define OFF_G1   2048                                            // 6 floats (softmaxed gates1)
#define OFF_G2   2080                                            // 6 floats
#define OFF_B1C  2176                                            // 256 floats combined bias1
#define OFF_B2C  3200                                            // 128 floats combined bias2
#define OFF_XH   4096                                            // f16 x: 16,777,216
#define OFF_W1C  (OFF_XH  + (size_t)NBATCH*T_LEN*C_MODEL*2)      // f16 [121][128][256]
#define OFF_W2C  (OFF_W1C + (size_t)RTAPS*C_MODEL*C_FF*2)        // f16 [121][256][128]
#define OFF_H    (OFF_W2C + (size_t)RTAPS*C_FF*C_MODEL*2)        // f16 hidden: 128*1056*256*2
// total ~ 101.9 MB

union ABreg { v16h v; uint4 q[2]; };

// ---------------------------------------------------------------------------
// Kernel 0: per-job metadata + gate softmax + combined biases.
// ---------------------------------------------------------------------------
__global__ __launch_bounds__(256)
void meta_kernel(const int* __restrict__ per, const float* __restrict__ wts,
                 const float* __restrict__ gates1, const float* __restrict__ gates2,
                 const float* b1_0, const float* b1_1, const float* b1_2,
                 const float* b1_3, const float* b1_4, const float* b1_5,
                 const float* b2_0, const float* b2_1, const float* b2_2,
                 const float* b2_3, const float* b2_4, const float* b2_5,
                 unsigned char* __restrict__ ws)
{
    int tid = threadIdx.x;

    float g1[6], g2[6];
    {
        float m1 = gates1[0], m2 = gates2[0];
        for (int i = 1; i < 6; ++i) { m1 = fmaxf(m1, gates1[i]); m2 = fmaxf(m2, gates2[i]); }
        float s1 = 0.f, s2 = 0.f;
        for (int i = 0; i < 6; ++i) {
            g1[i] = __expf(gates1[i] - m1); s1 += g1[i];
            g2[i] = __expf(gates2[i] - m2); s2 += g2[i];
        }
        for (int i = 0; i < 6; ++i) { g1[i] /= s1; g2[i] /= s2; }
    }
    float* gw1 = (float*)(ws + OFF_G1);
    float* gw2 = (float*)(ws + OFF_G2);
    if (tid < 6) { gw1[tid] = g1[tid]; gw2[tid] = g2[tid]; }

    const float* b1p[6] = { b1_0, b1_1, b1_2, b1_3, b1_4, b1_5 };
    const float* b2p[6] = { b2_0, b2_1, b2_2, b2_3, b2_4, b2_5 };
    if (tid < C_FF) {
        float a = 0.f;
        for (int i = 0; i < 6; ++i) a += g1[i] * b1p[i][tid];
        ((float*)(ws + OFF_B1C))[tid] = a;
    }
    if (tid < C_MODEL) {
        float a = 0.f;
        for (int i = 0; i < 6; ++i) a += g2[i] * b2p[i][tid];
        ((float*)(ws + OFF_B2C))[tid] = a;
    }

    if (tid < NJOBS) {
        int b = tid >> 1, slot = tid & 1;
        int p0 = per[b * 2 + 0];
        int p1 = per[b * 2 + 1];
        float u0 = wts[b * 2 + 0], u1 = wts[b * 2 + 1];
        float mu = fmaxf(u0, u1);
        float e0 = __expf(u0 - mu), e1 = __expf(u1 - mu);
        float inv = 1.f / (e0 + e1);
        float w0 = e0 * inv, w1 = e1 * inv;

        int p; float w; bool active;
        if (slot == 0) { p = p0; active = (p0 >= 1); w = w0 + ((p1 == p0) ? w1 : 0.f); }
        else           { p = p1; active = (p1 >= 1) && (p1 != p0); w = w1; }
        if (!active) p = 0;
        int P  = (p > 0) ? (T_LEN + p - 1) / p : 0;
        int TT = P * p;

        int*   mi = (int*)(ws + OFF_META);
        float* mf = (float*)(ws + OFF_META);
        mi[tid * 4 + 0] = p;
        mi[tid * 4 + 1] = P;
        mi[tid * 4 + 2] = TT;
        mf[tid * 4 + 3] = w;
    }
}

// ---------------------------------------------------------------------------
// Kernel 1: cast x (f32) -> f16 activations [b][t][c].
// ---------------------------------------------------------------------------
__global__ __launch_bounds__(256)
void cast_x_kernel(const float* __restrict__ x, _Float16* __restrict__ xh, int n)
{
    int i = blockIdx.x * 256 + threadIdx.x;
    if (i < n) xh[i] = (_Float16)x[i];
}

// ---------------------------------------------------------------------------
// Kernel 2: gate-combine six kxk kernels into one 11x11 f16 kernel,
// layout [tap r][cin][cout] (cout contiguous).
// ---------------------------------------------------------------------------
__global__ __launch_bounds__(256)
void combine_w_kernel(const float* W0, const float* W1, const float* W2,
                      const float* W3, const float* W4, const float* W5,
                      const float* __restrict__ g, _Float16* __restrict__ Wc,
                      int CIN, int COUT)
{
    long long idx   = (long long)blockIdx.x * 256 + threadIdx.x;
    long long total = (long long)RTAPS * CIN * COUT;
    if (idx >= total) return;

    int cout = (int)(idx % COUT);
    int cin  = (int)((idx / COUT) % CIN);
    int r    = (int)(idx / ((long long)COUT * CIN));
    int dy = r / 11, dx = r % 11;

    const float* Wp[6] = { W0, W1, W2, W3, W4, W5 };
    float acc = 0.f;
#pragma unroll
    for (int i = 0; i < 6; ++i) {
        int k  = 2 * i + 1;
        int iy = dy - 5 + i;
        int ix = dx - 5 + i;
        if ((unsigned)iy < (unsigned)k && (unsigned)ix < (unsigned)k)
            acc += g[i] * Wp[i][(((long long)cout * CIN + cin) * k + iy) * k + ix];
    }
    Wc[idx] = (_Float16)acc;
}

// ---------------------------------------------------------------------------
// Kernel 3: conv1 (128 -> 256) + bias + exact GELU -> f16 hidden.
// 512 threads = 16 waves; wave w owns t-rows [bx*256 + w*16, +16) and ALL 256
// couts (16 accumulator tiles). Per tap: 64KB weight slab staged in LDS and
// read by all 16 waves; next slab is fetched into registers during the WMMAs.
// WMMA per-lane layouts:
//   A (16x32 f16): lane L: row M=L&15; K halves {c..c+7},{c+16..c+23}, c=(L<16?0:8).
//   B (32x16 f16): lane L: col N=L&15; K halves {c..c+7},{c+16..c+23}, c=(L<16?0:8).
//   C (16x16 f32): vgpr r, lane L -> (M = r + (L>=16?8:0), N = L&15).
// ---------------------------------------------------------------------------
__global__ __launch_bounds__(512)
void conv1_gelu_kernel(unsigned char* __restrict__ ws)
{
    extern __shared__ _Float16 smem[];                 // 64KB: one tap slab [cin][cout]
    const int* mi = (const int*)(ws + OFF_META);
    int job = blockIdx.y;
    int p = mi[job * 4 + 0];
    if (p < 1) return;                                 // uniform: before any barrier
    int P  = mi[job * 4 + 1];
    int TT = mi[job * 4 + 2];
    int rows0 = blockIdx.x * 256;
    if (rows0 >= TT) return;                           // uniform

    int tid  = threadIdx.x;
    int lane = tid & 31;
    int wave = tid >> 5;
    int t0   = rows0 + wave * 16;
    int hi   = lane >> 4;
    int t    = t0 + (lane & 15);
    int row  = t / p;
    int col  = t - row * p;
    int khalfA = hi * 8;
    int kposB  = (lane & 15) + (hi << 4);

    const _Float16* Xb  = (const _Float16*)(ws + OFF_XH) + (size_t)(job >> 1) * T_LEN * C_MODEL;
    const uint4*    Wg  = (const uint4*)(ws + OFF_W1C);
    const float*    b1c = (const float*)(ws + OFF_B1C);
    _Float16*       H   = (_Float16*)(ws + OFF_H) + (size_t)job * HROWS * C_FF;

    v8f  acc[16] = {};
    uint4 stg[8];

    // stage tap 0
#pragma unroll
    for (int j = 0; j < 8; ++j) stg[j] = Wg[j * 512 + tid];
    {
        uint4* d = (uint4*)smem;
#pragma unroll
        for (int j = 0; j < 8; ++j) d[j * 512 + tid] = stg[j];
    }
    __syncthreads();

    int dy = 0, dx = 0;
    for (int r = 0; r < RTAPS; ++r) {
        if (r + 1 < RTAPS) {                           // overlap: fetch next slab into regs
            const uint4* s = Wg + (size_t)(r + 1) * SLAB_U4;
#pragma unroll
            for (int j = 0; j < 8; ++j) stg[j] = s[j * 512 + tid];
            if (r + 2 < RTAPS)
                __builtin_prefetch((const char*)(Wg + (size_t)(r + 2) * SLAB_U4) + tid * 128, 0, 1);
        }

        int rowp = row + dy - 5;
        int colp = col + dx - 5;
        int tp   = t + (dy - 5) * p + (dx - 5);        // == rowp*p + colp
        bool valid = (rowp >= 0) & (rowp < P) & (colp >= 0) & (colp < p) & (tp < T_LEN);
        const _Float16* arow = Xb + (size_t)tp * C_MODEL;

#pragma unroll
        for (int kc = 0; kc < 4; ++kc) {               // 128 cin in 4 chunks of 32
            ABreg a;
            if (valid) {
                const _Float16* ap = arow + kc * 32 + khalfA;
                a.q[0] = *(const uint4*)ap;
                a.q[1] = *(const uint4*)(ap + 16);
            } else {
                a.q[0] = make_uint4(0u, 0u, 0u, 0u);
                a.q[1] = make_uint4(0u, 0u, 0u, 0u);
            }
            const _Float16* bbase = smem + (size_t)(kc * 32 + khalfA) * C_FF + (lane & 15);
#pragma unroll
            for (int nt = 0; nt < 16; ++nt) {          // all 256 couts
                ABreg bm;
#pragma unroll
                for (int s = 0; s < 8; ++s) {
                    bm.v[s]     = bbase[s * C_FF + nt * 16];
                    bm.v[s + 8] = bbase[(s + 16) * C_FF + nt * 16];
                }
                acc[nt] = __builtin_amdgcn_wmma_f32_16x16x32_f16(
                    false, a.v, false, bm.v, (short)0, acc[nt], false, false);
            }
        }
        __syncthreads();                               // done reading slab r
        if (r + 1 < RTAPS) {
            uint4* d = (uint4*)smem;
#pragma unroll
            for (int j = 0; j < 8; ++j) d[j * 512 + tid] = stg[j];
            __syncthreads();                           // slab r+1 visible
        }
        if (++dx == 11) { dx = 0; ++dy; }
    }

    // epilogue: bias + exact GELU, store f16 hidden
#pragma unroll
    for (int nt = 0; nt < 16; ++nt) {
        int n = nt * 16 + (lane & 15);
        float bias = b1c[n];
#pragma unroll
        for (int r8 = 0; r8 < 8; ++r8) {
            int to = t0 + r8 + hi * 8;
            if (to < TT) {
                float v = acc[nt][r8] + bias;
                float gel = 0.5f * v * (1.0f + erff(v * 0.70710678118654752440f));
                H[(size_t)to * C_FF + n] = (_Float16)gel;
            }
        }
    }
}

// ---------------------------------------------------------------------------
// Kernel 4: conv2 (256 -> 128), both period slots accumulated in-wave against
// ONE LDS weight staging (weights shared by slots), + bias, * period weight,
// fused residual +x, f32 output. 512 threads = 16 waves; wave w owns t-rows
// [bx*256 + w*16, +16) and all 128 couts for both slots.
// ---------------------------------------------------------------------------
__global__ __launch_bounds__(512)
void conv2_out_kernel(const float* __restrict__ x, float* __restrict__ out,
                      unsigned char* __restrict__ ws)
{
    extern __shared__ _Float16 smem[];                 // 64KB: one tap slab [cin=256][cout=128]
    int b   = blockIdx.y;
    int tid = threadIdx.x;
    int lane = tid & 31;
    int wave = tid >> 5;
    int t0  = blockIdx.x * 256 + wave * 16;
    int hi  = lane >> 4;
    int t   = t0 + (lane & 15);
    int khalfA = hi * 8;
    int kposB  = (lane & 15) + (hi << 4);

    const int*   mi = (const int*)(ws + OFF_META);
    const float* mf = (const float*)(ws + OFF_META);
    int   pa = mi[(b * 2 + 0) * 4 + 0], Pa = mi[(b * 2 + 0) * 4 + 1];
    float wa = mf[(b * 2 + 0) * 4 + 3];
    int   pb = mi[(b * 2 + 1) * 4 + 0], Pb = mi[(b * 2 + 1) * 4 + 1];
    float wb = mf[(b * 2 + 1) * 4 + 3];
    bool act0 = (pa >= 1), act1 = (pb >= 1);

    if (!act0 && !act1) {                              // uniform: residual only
#pragma unroll
        for (int nt = 0; nt < 8; ++nt) {
            int n = nt * 16 + (lane & 15);
#pragma unroll
            for (int r8 = 0; r8 < 8; ++r8) {
                size_t idx = ((size_t)b * T_LEN + t0 + r8 + hi * 8) * C_MODEL + n;
                out[idx] = x[idx];
            }
        }
        return;
    }

    int pp0 = act0 ? pa : 1, pp1 = act1 ? pb : 1;
    int row0 = t / pp0, col0 = t - row0 * pp0;
    int row1 = t / pp1, col1 = t - row1 * pp1;
    const _Float16* H0 = (const _Float16*)(ws + OFF_H) + (size_t)(b * 2 + 0) * HROWS * C_FF;
    const _Float16* H1 = (const _Float16*)(ws + OFF_H) + (size_t)(b * 2 + 1) * HROWS * C_FF;
    const uint4*    Wg  = (const uint4*)(ws + OFF_W2C);
    const float*    b2c = (const float*)(ws + OFF_B2C);

    v8f  acc0[8] = {}, acc1[8] = {};
    uint4 stg[8];

#pragma unroll
    for (int j = 0; j < 8; ++j) stg[j] = Wg[j * 512 + tid];
    {
        uint4* d = (uint4*)smem;
#pragma unroll
        for (int j = 0; j < 8; ++j) d[j * 512 + tid] = stg[j];
    }
    __syncthreads();

    int dy = 0, dx = 0;
    for (int r = 0; r < RTAPS; ++r) {
        if (r + 1 < RTAPS) {
            const uint4* s = Wg + (size_t)(r + 1) * SLAB_U4;
#pragma unroll
            for (int j = 0; j < 8; ++j) stg[j] = s[j * 512 + tid];
            if (r + 2 < RTAPS)
                __builtin_prefetch((const char*)(Wg + (size_t)(r + 2) * SLAB_U4) + tid * 128, 0, 1);
        }

        if (act0) {
            int rowp = row0 + dy - 5;
            int colp = col0 + dx - 5;
            int tp   = t + (dy - 5) * pp0 + (dx - 5);
            bool valid = (rowp >= 0) & (rowp < Pa) & (colp >= 0) & (colp < pp0);
            const _Float16* arow = H0 + (size_t)tp * C_FF;
#pragma unroll
            for (int kc = 0; kc < 8; ++kc) {
                ABreg a;
                if (valid) {
                    const _Float16* ap = arow + kc * 32 + khalfA;
                    a.q[0] = *(const uint4*)ap;
                    a.q[1] = *(const uint4*)(ap + 16);
                } else {
                    a.q[0] = make_uint4(0u, 0u, 0u, 0u);
                    a.q[1] = make_uint4(0u, 0u, 0u, 0u);
                }
                const _Float16* bbase = smem + (size_t)(kc * 32 + khalfA) * C_MODEL + (lane & 15);
#pragma unroll
                for (int nt = 0; nt < 8; ++nt) {
                    ABreg bm;
#pragma unroll
                    for (int s = 0; s < 8; ++s) {
                        bm.v[s]     = bbase[s * C_MODEL + nt * 16];
                        bm.v[s + 8] = bbase[(s + 16) * C_MODEL + nt * 16];
                    }
                    acc0[nt] = __builtin_amdgcn_wmma_f32_16x16x32_f16(
                        false, a.v, false, bm.v, (short)0, acc0[nt], false, false);
                }
            }
        }
        if (act1) {
            int rowp = row1 + dy - 5;
            int colp = col1 + dx - 5;
            int tp   = t + (dy - 5) * pp1 + (dx - 5);
            bool valid = (rowp >= 0) & (rowp < Pb) & (colp >= 0) & (colp < pp1);
            const _Float16* arow = H1 + (size_t)tp * C_FF;
#pragma unroll
            for (int kc = 0; kc < 8; ++kc) {
                ABreg a;
                if (valid) {
                    const _Float16* ap = arow + kc * 32 + khalfA;
                    a.q[0] = *(const uint4*)ap;
                    a.q[1] = *(const uint4*)(ap + 16);
                } else {
                    a.q[0] = make_uint4(0u, 0u, 0u, 0u);
                    a.q[1] = make_uint4(0u, 0u, 0u, 0u);
                }
                const _Float16* bbase = smem + (size_t)(kc * 32 + khalfA) * C_MODEL + (lane & 15);
#pragma unroll
                for (int nt = 0; nt < 8; ++nt) {
                    ABreg bm;
#pragma unroll
                    for (int s = 0; s < 8; ++s) {
                        bm.v[s]     = bbase[s * C_MODEL + nt * 16];
                        bm.v[s + 8] = bbase[(s + 16) * C_MODEL + nt * 16];
                    }
                    acc1[nt] = __builtin_amdgcn_wmma_f32_16x16x32_f16(
                        false, a.v, false, bm.v, (short)0, acc1[nt], false, false);
                }
            }
        }
        __syncthreads();
        if (r + 1 < RTAPS) {
            uint4* d = (uint4*)smem;
#pragma unroll
            for (int j = 0; j < 8; ++j) d[j * 512 + tid] = stg[j];
            __syncthreads();
        }
        if (++dx == 11) { dx = 0; ++dy; }
    }

    // epilogue: per-slot (acc + bias) * w, residual, store f32
#pragma unroll
    for (int nt = 0; nt < 8; ++nt) {
        int n = nt * 16 + (lane & 15);
        float bias = b2c[n];
#pragma unroll
        for (int r8 = 0; r8 < 8; ++r8) {
            float tot = 0.f;
            if (act0) tot += wa * (acc0[nt][r8] + bias);
            if (act1) tot += wb * (acc1[nt][r8] + bias);
            size_t idx = ((size_t)b * T_LEN + t0 + r8 + hi * 8) * C_MODEL + n;
            out[idx] = x[idx] + tot;
        }
    }
}

// ---------------------------------------------------------------------------
// Launch. Input order (setup_inputs dict, tuples flattened):
//   0:x  1:per_tensor  2:weights  3..8:W1[6]  9..14:b1[6]  15:gates1
//   16..21:W2[6]  22..27:b2[6]  28:gates2
// per_tensor read as int32 (JAX default x64-disabled stores int32).
// Needs ws_size >= ~102 MB.
// ---------------------------------------------------------------------------
extern "C" void kernel_launch(void* const* d_in, const int* in_sizes, int n_in,
                              void* d_out, int out_size, void* d_ws, size_t ws_size,
                              hipStream_t stream)
{
    (void)in_sizes; (void)n_in; (void)out_size; (void)ws_size;

    const float* x      = (const float*)d_in[0];
    const int*   per    = (const int*)d_in[1];
    const float* wts    = (const float*)d_in[2];
    const float* W1p[6], *b1p[6], *W2p[6], *b2p[6];
    for (int i = 0; i < 6; ++i) {
        W1p[i] = (const float*)d_in[3 + i];
        b1p[i] = (const float*)d_in[9 + i];
        W2p[i] = (const float*)d_in[16 + i];
        b2p[i] = (const float*)d_in[22 + i];
    }
    const float* gates1 = (const float*)d_in[15];
    const float* gates2 = (const float*)d_in[28];
    unsigned char* ws   = (unsigned char*)d_ws;
    float* out          = (float*)d_out;

    meta_kernel<<<1, 256, 0, stream>>>(per, wts, gates1, gates2,
        b1p[0], b1p[1], b1p[2], b1p[3], b1p[4], b1p[5],
        b2p[0], b2p[1], b2p[2], b2p[3], b2p[4], b2p[5], ws);

    int nx = NBATCH * T_LEN * C_MODEL;
    cast_x_kernel<<<(nx + 255) / 256, 256, 0, stream>>>(x, (_Float16*)(ws + OFF_XH), nx);

    long long tot = (long long)RTAPS * C_MODEL * C_FF;
    int cgrid = (int)((tot + 255) / 256);
    combine_w_kernel<<<cgrid, 256, 0, stream>>>(
        W1p[0], W1p[1], W1p[2], W1p[3], W1p[4], W1p[5],
        (const float*)(ws + OFF_G1), (_Float16*)(ws + OFF_W1C), C_MODEL, C_FF);
    combine_w_kernel<<<cgrid, 256, 0, stream>>>(
        W2p[0], W2p[1], W2p[2], W2p[3], W2p[4], W2p[5],
        (const float*)(ws + OFF_G2), (_Float16*)(ws + OFF_W2C), C_FF, C_MODEL);

    size_t slab_bytes = (size_t)C_MODEL * C_FF * sizeof(_Float16);   // 64KB dynamic LDS
    conv1_gelu_kernel<<<dim3(5, NJOBS), 512, slab_bytes, stream>>>(ws);
    conv2_out_kernel<<<dim3(4, NBATCH), 512, slab_bytes, stream>>>(x, out, ws);
}
